// MSDeformAttn_20169166422030
// MI455X (gfx1250) — compile-verified, matrix operand order checked by reading, not run on hardware
//
#include <hip/hip_runtime.h>
#include <hip/hip_bf16.h>

typedef __attribute__((ext_vector_type(16))) _Float16 v16h;
typedef __attribute__((ext_vector_type(8)))  _Float16 v8h;
typedef __attribute__((ext_vector_type(8)))  float    v8f;

#define D_MODEL 256
#define NH      8
#define NL      4
#define NP      4
#define HD      32
#define LQ      5440
#define LV      5440
#define BATCH   4

// ---------------------------------------------------------------------------
// Weight prep: Wt[n*256 + k] = (f16) W[k*N + n].  (K is always 256 here.)
// Tiny (<=256x256); one-time cost, makes every GEMM B-fragment two aligned
// 16B loads instead of 16 strided dword loads.
// ---------------------------------------------------------------------------
__global__ __launch_bounds__(256) void wt_transpose_f16(
    const float* __restrict__ W, _Float16* __restrict__ Wt, int N)
{
    const int idx = blockIdx.x * 256 + threadIdx.x;   // over 256*N elements
    if (idx >= 256 * N) return;
    const int k = idx / N;
    const int n = idx - k * N;
    Wt[n * 256 + k] = (_Float16)W[idx];
}

// Elementwise f32 -> f16 (activations), 4 elems/thread, coalesced.
__global__ __launch_bounds__(256) void cvt_f32_to_f16(
    const float* __restrict__ src, _Float16* __restrict__ dst, int n4)
{
    const int i = blockIdx.x * 256 + threadIdx.x;     // over n/4 quads
    if (i >= n4) return;
    const float4 q = ((const float4*)src)[i];
    v8h unused; (void)unused;
    dst[i * 4 + 0] = (_Float16)q.x;
    dst[i * 4 + 1] = (_Float16)q.y;
    dst[i * 4 + 2] = (_Float16)q.z;
    dst[i * 4 + 3] = (_Float16)q.w;
}

// ---------------------------------------------------------------------------
// GEMM: out[M,N] = A[M,256](f16) x Wt[N,256](f16, pre-transposed) + bias[N].
// One wave32 computes a 64x16 tile (4 M-subtiles reuse one B fragment per
// k-step -> 4 v_wmma_f32_16x16x32_f16 per 10 b128 loads). 8 waves/block.
// A/B fragment layout (ISA 7.12.2, 16-bit 16x32): lane l -> row l&15,
// kg = l>>4; frag elem i covers K = (i<8?0:16) + 8*kg + (i&7).
// C/D layout: VGPR r, lane l -> M = r + 8*(l>>4), N = l&15.
// All flat indices < 2^31 -> 32-bit address math.
// ---------------------------------------------------------------------------
__global__ __launch_bounds__(256) void gemm_f16wmma_bias(
    const _Float16* __restrict__ A, const _Float16* __restrict__ Wt,
    const float* __restrict__ bias, float* __restrict__ out,
    int M, int N)
{
    const int lane    = threadIdx.x & 31;
    const int waveId  = threadIdx.x >> 5;
    const int tilesN  = N >> 4;
    const int tileId  = blockIdx.x * 8 + waveId;
    const int numTile = (M >> 6) * tilesN;            // 64-row M strips
    if (tileId >= numTile) return;                    // wave-uniform exit

    const int tileM    = tileId / tilesN;
    const int tileN    = tileId - tileM * tilesN;
    const int halfLane = lane & 15;
    const int kg       = lane >> 4;
    const int rowBase  = tileM * 64 + halfLane;
    const int col      = tileN * 16 + halfLane;

    v8f c0 = {0.f,0.f,0.f,0.f,0.f,0.f,0.f,0.f};
    v8f c1 = c0, c2 = c0, c3 = c0;

    const _Float16* __restrict__ wrow = Wt + col * 256;
    const _Float16* __restrict__ a0   = A + (rowBase +  0) * 256;
    const _Float16* __restrict__ a1   = A + (rowBase + 16) * 256;
    const _Float16* __restrict__ a2   = A + (rowBase + 32) * 256;
    const _Float16* __restrict__ a3   = A + (rowBase + 48) * 256;

    #pragma unroll
    for (int k0 = 0; k0 < 256; k0 += 32) {
        const int ko = k0 + kg * 8;                   // 16B-aligned half offset

        const v8h bl = *(const v8h*)(wrow + ko);
        const v8h bh = *(const v8h*)(wrow + ko + 16);
        const v16h bf = __builtin_shufflevector(bl, bh,
                            0,1,2,3,4,5,6,7,8,9,10,11,12,13,14,15);

        const v8h l0 = *(const v8h*)(a0 + ko), h0 = *(const v8h*)(a0 + ko + 16);
        const v8h l1 = *(const v8h*)(a1 + ko), h1 = *(const v8h*)(a1 + ko + 16);
        const v8h l2 = *(const v8h*)(a2 + ko), h2 = *(const v8h*)(a2 + ko + 16);
        const v8h l3 = *(const v8h*)(a3 + ko), h3 = *(const v8h*)(a3 + ko + 16);
        const v16h f0 = __builtin_shufflevector(l0, h0, 0,1,2,3,4,5,6,7,8,9,10,11,12,13,14,15);
        const v16h f1 = __builtin_shufflevector(l1, h1, 0,1,2,3,4,5,6,7,8,9,10,11,12,13,14,15);
        const v16h f2 = __builtin_shufflevector(l2, h2, 0,1,2,3,4,5,6,7,8,9,10,11,12,13,14,15);
        const v16h f3 = __builtin_shufflevector(l3, h3, 0,1,2,3,4,5,6,7,8,9,10,11,12,13,14,15);

        c0 = __builtin_amdgcn_wmma_f32_16x16x32_f16(false, f0, false, bf, (short)0, c0, false, false);
        c1 = __builtin_amdgcn_wmma_f32_16x16x32_f16(false, f1, false, bf, (short)0, c1, false, false);
        c2 = __builtin_amdgcn_wmma_f32_16x16x32_f16(false, f2, false, bf, (short)0, c2, false, false);
        c3 = __builtin_amdgcn_wmma_f32_16x16x32_f16(false, f3, false, bf, (short)0, c3, false, false);
    }

    const float bv    = bias[col];
    const int   mBase = tileM * 64 + (kg << 3);
    #pragma unroll
    for (int r = 0; r < 8; ++r) {
        out[(mBase +  0 + r) * N + col] = c0[r] + bv;
        out[(mBase + 16 + r) * N + col] = c1[r] + bv;
        out[(mBase + 32 + r) * N + col] = c2[r] + bv;
        out[(mBase + 48 + r) * N + col] = c3[r] + bv;
    }
}

// ---------------------------------------------------------------------------
// Softmax over 16 logits per (b, q, head), in place.
// ---------------------------------------------------------------------------
__global__ __launch_bounds__(256) void softmax16(float* __restrict__ attn, int rows)
{
    const int t = blockIdx.x * 256 + threadIdx.x;
    if (t >= rows) return;
    float* p = attn + t * 16;
    float v[16];
    float mx = -3.0e38f;
    #pragma unroll
    for (int i = 0; i < 16; ++i) { v[i] = p[i]; mx = fmaxf(mx, v[i]); }
    float s = 0.f;
    #pragma unroll
    for (int i = 0; i < 16; ++i) { v[i] = __expf(v[i] - mx); s += v[i]; }
    const float inv = 1.f / s;
    #pragma unroll
    for (int i = 0; i < 16; ++i) p[i] = v[i] * inv;
}

// ---------------------------------------------------------------------------
// Deformable bilinear sampling + weighted accumulation.
// One wave32 per (b, q, head); lane = channel (HD == 32 == wave32 width).
// Sampling coords are wave-uniform -> non-divergent validity branches; the
// projected value tensor (22 MB) is L2-resident, so corner gathers hit L2.
// Emits the accumulator directly as f16 in [B*Lq, 256] layout so the output
// projection is the same WMMA GEMM.
// ---------------------------------------------------------------------------
__global__ __launch_bounds__(256) void msda_sample(
    const float* __restrict__ valp,    // [B*Lv, 256] f32 (c = head*32 + chan)
    const float* __restrict__ refp,    // [B, Lq, NL, 2]
    const float* __restrict__ offs,    // [B*Lq, 256] = (nh, nl, np, 2)
    const float* __restrict__ attn,    // [B*Lq, 128] = (nh, nl*np), softmaxed
    _Float16* __restrict__ acc_out)    // [B*Lq, 256] f16
{
    const int lane = threadIdx.x & 31;
    const int wave = (blockIdx.x * 256 + threadIdx.x) >> 5;
    if (wave >= BATCH * LQ * NH) return;

    const int h  = wave % NH;
    const int bq = wave / NH;
    const int b  = bq / LQ;

    const int HARR[NL]  = {64, 32, 16, 8};
    const int WARR[NL]  = {64, 32, 16, 8};
    const int START[NL] = {0, 4096, 5120, 5376};

    const float* __restrict__ vbase = valp + b * LV * D_MODEL + h * HD + lane;
    const float* __restrict__ rp    = refp + bq * (NL * 2);
    const float* __restrict__ op    = offs + bq * D_MODEL + h * (NL * NP * 2);
    const float* __restrict__ wp    = attn + bq * (NH * NL * NP) + h * (NL * NP);

    float acc = 0.f;
    #pragma unroll
    for (int lid = 0; lid < NL; ++lid) {
        const int   Hh = HARR[lid], Ww = WARR[lid];
        const float fH = (float)Hh, fW = (float)Ww;
        const float rx = rp[lid * 2 + 0];
        const float ry = rp[lid * 2 + 1];
        const float* __restrict__ vlvl = vbase + START[lid] * D_MODEL;

        #pragma unroll
        for (int p = 0; p < NP; ++p) {
            const float ox = op[lid * (NP * 2) + p * 2 + 0];
            const float oy = op[lid * (NP * 2) + p * 2 + 1];
            // ((ref + off/W)*2 - 1 + 1)*0.5*W - 0.5  ==  ref*W + off - 0.5
            const float x = rx * fW + ox - 0.5f;
            const float y = ry * fH + oy - 0.5f;
            const float xf = floorf(x), yf = floorf(y);
            const float wx = x - xf,    wy = y - yf;
            const int   x0 = (int)xf,   y0 = (int)yf;

            auto corner = [&](int xi, int yi) -> float {
                if (xi < 0 || xi >= Ww || yi < 0 || yi >= Hh) return 0.f;
                return vlvl[(yi * Ww + xi) * D_MODEL];
            };
            const float s =
                  corner(x0,     y0    ) * (1.f - wx) * (1.f - wy)
                + corner(x0 + 1, y0    ) * wx         * (1.f - wy)
                + corner(x0,     y0 + 1) * (1.f - wx) * wy
                + corner(x0 + 1, y0 + 1) * wx         * wy;

            acc += wp[lid * NP + p] * s;
        }
    }
    acc_out[bq * D_MODEL + h * HD + lane] = (_Float16)acc;
}

// ---------------------------------------------------------------------------
extern "C" void kernel_launch(void* const* d_in, const int* in_sizes, int n_in,
                              void* d_out, int out_size, void* d_ws, size_t ws_size,
                              hipStream_t stream)
{
    (void)in_sizes; (void)n_in; (void)out_size; (void)ws_size;

    const float* query  = (const float*)d_in[0];
    const float* refp   = (const float*)d_in[1];
    const float* value  = (const float*)d_in[2];
    const float* W_off  = (const float*)d_in[3];
    const float* b_off  = (const float*)d_in[4];
    const float* W_attn = (const float*)d_in[5];
    const float* b_attn = (const float*)d_in[6];
    const float* W_val  = (const float*)d_in[7];
    const float* b_val  = (const float*)d_in[8];
    const float* W_out  = (const float*)d_in[9];
    const float* b_out  = (const float*)d_in[10];
    float* out = (float*)d_out;

    const int rows = BATCH * LQ;                        // 21760 = 340 * 64

    // ---- workspace carve-up ----
    char* ws = (char*)d_ws;
    float*    val_proj = (float*)ws;                     ws += (size_t)rows * 256 * 4;
    float*    off_buf  = (float*)ws;                     ws += (size_t)rows * 256 * 4;
    float*    attn_buf = (float*)ws;                     ws += (size_t)rows * 128 * 4;
    _Float16* query_h  = (_Float16*)ws;                  ws += (size_t)rows * 256 * 2;
    _Float16* value_h  = (_Float16*)ws;                  ws += (size_t)rows * 256 * 2;
    _Float16* acc_h    = (_Float16*)ws;                  ws += (size_t)rows * 256 * 2;
    _Float16* Wt_val   = (_Float16*)ws;                  ws += 256 * 256 * 2;
    _Float16* Wt_off   = (_Float16*)ws;                  ws += 256 * 256 * 2;
    _Float16* Wt_attn  = (_Float16*)ws;                  ws += 256 * 128 * 2;
    _Float16* Wt_out   = (_Float16*)ws;

    const dim3 blk(256);

    // ---- one-time weight transpose/convert (tiny) ----
    wt_transpose_f16<<<256, blk, 0, stream>>>(W_val,  Wt_val,  256);
    wt_transpose_f16<<<256, blk, 0, stream>>>(W_off,  Wt_off,  256);
    wt_transpose_f16<<<128, blk, 0, stream>>>(W_attn, Wt_attn, 128);
    wt_transpose_f16<<<256, blk, 0, stream>>>(W_out,  Wt_out,  256);

    // ---- activation f32 -> f16 ----
    const int n4 = rows * 256 / 4;                      // 1,392,640 quads
    cvt_f32_to_f16<<<(n4 + 255) / 256, blk, 0, stream>>>(query, query_h, n4);
    cvt_f32_to_f16<<<(n4 + 255) / 256, blk, 0, stream>>>(value, value_h, n4);

    // ---- projections (WMMA) ----
    const int t256 = (rows / 64) * (256 / 16);          // 5440 -> 680 blocks
    const int t128 = (rows / 64) * (128 / 16);          // 2720 -> 340 blocks
    gemm_f16wmma_bias<<<(t256 + 7) / 8, blk, 0, stream>>>(value_h, Wt_val,  b_val,  val_proj, rows, 256);
    gemm_f16wmma_bias<<<(t256 + 7) / 8, blk, 0, stream>>>(query_h, Wt_off,  b_off,  off_buf,  rows, 256);
    gemm_f16wmma_bias<<<(t128 + 7) / 8, blk, 0, stream>>>(query_h, Wt_attn, b_attn, attn_buf, rows, 128);

    // ---- softmax over 16 sampling weights per (b,q,head) ----
    softmax16<<<(rows * NH + 255) / 256, blk, 0, stream>>>(attn_buf, rows * NH);

    // ---- deformable bilinear gather/accumulate (VALU, L2-resident) ----
    msda_sample<<<(rows * NH * 32 + 255) / 256, blk, 0, stream>>>(
        val_proj, refp, off_buf, attn_buf, acc_h);

    // ---- output projection (WMMA) ----
    gemm_f16wmma_bias<<<(t256 + 7) / 8, blk, 0, stream>>>(acc_h, Wt_out, b_out, out, rows, 256);
}
